// GraphEncoder_37022618091730
// MI455X (gfx1250) — matrix operand order repaired and checked by reading, not compiled, not run on hardware
//
#include <hip/hip_runtime.h>
#include <hip/hip_bf16.h>
#include <stdint.h>

// Problem constants (match reference)
#define NN    25000
#define EE    200000
#define MPAD  25008          // NN rounded up to multiple of 16
#define TILE_E 32            // edges per workgroup in fused edge-MLP kernel

typedef __attribute__((ext_vector_type(16))) __bf16       v16bf;
typedef __attribute__((ext_vector_type(8)))  float        v8f;
typedef __attribute__((ext_vector_type(8)))  unsigned int v8u;

union FragU { v8u u; v16bf h; };

__device__ __forceinline__ unsigned short f2bf(float f) {
  unsigned int u = __float_as_uint(f);
  u += 0x7FFFu + ((u >> 16) & 1u);           // round-to-nearest-even
  return (unsigned short)(u >> 16);
}
__device__ __forceinline__ float eluf(float x)  { return x > 0.f ? x : (__expf(x) - 1.f); }
__device__ __forceinline__ float lrelu(float x) { return x > 0.f ? x : 0.2f * x; }

__device__ __forceinline__ v8f wmma_bf16(v16bf a, v16bf b, v8f c) {
  // D = A(16x32 bf16) x B(32x16 bf16) + C(16x16 f32)
  return __builtin_amdgcn_wmma_f32_16x16x32_bf16(false, a, false, b, (short)0, c, false, false);
}

// A fragment (16x32 bf16) from a row-major bf16 buffer (LDS or global).
// ISA layout: lanes 0-15 -> M=0..15, VGPR j<4: K=2j + 8*grp, j>=4: K=16+2(j-4)+8*grp.
__device__ __forceinline__ v16bf load_a_frag(const unsigned short* base, int ld, int lane) {
  FragU f;
  const int grp = (lane >> 4) & 1;
  const unsigned short* row = base + (lane & 15) * ld;
#pragma unroll
  for (int j = 0; j < 8; ++j) {
    int kb = (j < 4) ? (2*j + 8*grp) : (16 + 2*(j-4) + 8*grp);
    f.u[j] = *(const unsigned int*)(row + kb);     // 4B aligned: kb even, ld even
  }
  return f.h;
}

// B fragment (32x16 bf16) from pre-packed fragment stream: tile-major, lane-major,
// 8 contiguous dwords per lane.
__device__ __forceinline__ v16bf load_b_frag(const unsigned int* pack, int tile, int lane) {
  FragU f;
  const unsigned int* p = pack + ((size_t)tile * 32 + lane) * 8;
#pragma unroll
  for (int j = 0; j < 8; ++j) f.u[j] = p[j];
  return f.h;
}

// ---------------- weight packing: row-major fp32 [K,NC] -> B fragments (bf16) -----
__global__ void k_pack_b(const float* __restrict__ W, unsigned int* __restrict__ out,
                         int K, int NC) {
  int idx = blockIdx.x * blockDim.x + threadIdx.x;   // one thread per (tile,lane)
  int ntn = NC >> 4;
  int total = (K >> 5) * ntn * 32;
  if (idx >= total) return;
  int lane = idx & 31, tile = idx >> 5;
  int kt = tile / ntn, nt = tile % ntn;
  int grp = lane >> 4, n = nt * 16 + (lane & 15);
  unsigned int* o = out + (size_t)idx * 8;
#pragma unroll
  for (int j = 0; j < 8; ++j) {
    int kk = kt * 32 + ((j < 4) ? (2*j + 8*grp) : (16 + 2*(j-4) + 8*grp));  // B layout: K = 2j+8*grp (j<4), 16+2(j-4)+8*grp (j>=4)
    unsigned int h0 = f2bf(W[(size_t)kk * NC + n]);
    unsigned int h1 = f2bf(W[(size_t)(kk + 1) * NC + n]);
    o[j] = h0 | (h1 << 16);
  }
}

// ---------------- generic fill ----------------------------------------------------
__global__ void k_fill(unsigned int* __restrict__ p, unsigned int v, size_t n) {
  size_t i = (size_t)blockIdx.x * blockDim.x + threadIdx.x;
  if (i < n) p[i] = v;
}

// ---------------- fused NNConv edge pipeline --------------------------------------
// Per block: 32 edges. h1=elu(attr@w1+b1) -> LDS bf16; h2=elu(h1@w2+b2) via WMMA
// -> LDS bf16; then for each i<16: w_i = elu(h2@w3[:,i*64:+64]+b3) via WMMA,
// msg += x[src][i]*w_i ; finally atomicAdd into node accumulator (segment_sum).
__global__ __launch_bounds__(256) void k_edge_fused(
    const float* __restrict__ xin,            // [NN,16]
    const int*   __restrict__ eidx,           // [2,EE]
    const float* __restrict__ attr,           // [EE,5]
    const float* __restrict__ w1, const float* __restrict__ b1,
    const unsigned int* __restrict__ w2p, const float* __restrict__ b2,
    const unsigned int* __restrict__ w3p, const float* __restrict__ b3,
    float* __restrict__ accn)                 // [NN,64]
{
  __shared__ unsigned short s_h1[TILE_E][256 + 8];
  __shared__ unsigned short s_h2[TILE_E][1024 + 8];
  __shared__ float s_x[TILE_E][16];
  __shared__ float s_attr[TILE_E][5];
  __shared__ int   s_dst[TILE_E];

  const int tid  = threadIdx.x;
  const int lane = tid & 31;
  const int wave = tid >> 5;
  const int e0   = blockIdx.x * TILE_E;

  // Phase 0: stage edge attrs, dst, gathered x[src]
  if (tid < TILE_E) s_dst[tid] = eidx[EE + e0 + tid];
  for (int i = tid; i < TILE_E * 5; i += 256)
    s_attr[i / 5][i % 5] = attr[(size_t)(e0 + i / 5) * 5 + (i % 5)];
  for (int i = tid; i < TILE_E * 16; i += 256) {
    int srcn = eidx[e0 + i / 16];
    s_x[i / 16][i % 16] = xin[(size_t)srcn * 16 + (i % 16)];
  }
  __syncthreads();

  // Phase 0b: h1 = elu(attr @ w1 + b1)   (K=5, plain VALU; one column per thread)
  {
    int c = tid;                        // 0..255
    float w1c[5];
#pragma unroll
    for (int k = 0; k < 5; ++k) w1c[k] = w1[k * 256 + c];
    float bc = b1[c];
    for (int e = 0; e < TILE_E; ++e) {
      float s = bc;
#pragma unroll
      for (int k = 0; k < 5; ++k) s += s_attr[e][k] * w1c[k];
      s_h1[e][c] = f2bf(eluf(s));
    }
  }
  __syncthreads();

  // Phase 1: h2 = elu(h1 @ w2 + b2); output [32,1024] = 2 x 64 tiles, 8 waves
  for (int t = wave; t < 2 * 64; t += 8) {
    int mt = t & 1, nt = t >> 1;
    v8f acc = {};
#pragma unroll
    for (int kt = 0; kt < 8; ++kt) {
      v16bf a = load_a_frag(&s_h1[mt * 16][kt * 32], 256 + 8, lane);
      v16bf b = load_b_frag(w2p, kt * 64 + nt, lane);
      acc = wmma_bf16(a, b, acc);
    }
    int col = nt * 16 + (lane & 15);
    float bb = b2[col];
    int grp = lane >> 4;
#pragma unroll
    for (int r = 0; r < 8; ++r)
      s_h2[mt * 16 + r + 8 * grp][col] = f2bf(eluf(acc[r] + bb));
  }
  __syncthreads();

  // Phase 2: fused GEMM3 + einsum. Each wave owns one fixed 16x16 msg tile.
  {
    const int mt = wave >> 2, nt = wave & 3;       // 2 x 4 tiles = 8 waves
    const int grp = lane >> 4, ncol = lane & 15;
    float msg[8] = {0.f, 0.f, 0.f, 0.f, 0.f, 0.f, 0.f, 0.f};
    for (int i = 0; i < 16; ++i) {                 // F_IN slices of w3 columns
      v8f acc = {};
      for (int kt = 0; kt < 32; ++kt) {
        v16bf a = load_a_frag(&s_h2[mt * 16][kt * 32], 1024 + 8, lane);
        v16bf b = load_b_frag(w3p, kt * 64 + (i * 4 + nt), lane);
        acc = wmma_bf16(a, b, acc);
      }
      int col = i * 64 + nt * 16 + ncol;
      float bb = b3[col];
#pragma unroll
      for (int r = 0; r < 8; ++r) {
        int row = mt * 16 + r + 8 * grp;
        msg[r] += s_x[row][i] * eluf(acc[r] + bb);
      }
    }
#pragma unroll
    for (int r = 0; r < 8; ++r) {
      int row = mt * 16 + r + 8 * grp;
      atomicAdd(&accn[(size_t)s_dst[row] * 64 + nt * 16 + ncol], msg[r]);
    }
  }
}

// ---------------- NNConv epilogue: x1 = accn + x @ root + nn_bias -----------------
__global__ void k_root(const float* __restrict__ xin, const float* __restrict__ root,
                       const float* __restrict__ nnb, const float* __restrict__ accn,
                       float* __restrict__ xout) {
  int idx = blockIdx.x * blockDim.x + threadIdx.x;
  if (idx >= NN * 64) return;
  int n = idx >> 6, o = idx & 63;
  float s = accn[idx] + nnb[o];
#pragma unroll
  for (int k = 0; k < 16; ++k) s += xin[(size_t)n * 16 + k] * root[k * 64 + o];
  xout[idx] = s;
}

// ---------------- fp32 -> bf16 (zero-padded to MPAD rows) -------------------------
__global__ void k_to_bf16(const float* __restrict__ X, unsigned short* __restrict__ O,
                          int nrows, int K) {
  int idx = blockIdx.x * blockDim.x + threadIdx.x;
  if (idx >= MPAD * K) return;
  int r = idx / K;
  O[idx] = (r < nrows) ? f2bf(X[idx]) : (unsigned short)0;
}

// ---------------- node GEMM: hx = A_bf16[MPAD,K] @ Bpack -> C f32[MPAD,NC] --------
__global__ __launch_bounds__(256) void k_gemm_nodes(
    const unsigned short* __restrict__ A, const unsigned int* __restrict__ Bp,
    float* __restrict__ C, int K, int NC) {
  const int wave = threadIdx.x >> 5, lane = threadIdx.x & 31;
  const int ntn = NC >> 4;
  const int total = (MPAD >> 4) * ntn;
  const int tile = blockIdx.x * 8 + wave;          // wave-uniform
  if (tile >= total) return;
  const int mt = tile / ntn, nt = tile % ntn;
  v8f acc = {};
  const int kts = K >> 5;
  for (int kt = 0; kt < kts; ++kt) {
    v16bf a = load_a_frag(A + (size_t)(mt * 16) * K + kt * 32, K, lane);
    v16bf b = load_b_frag(Bp, kt * ntn + nt, lane);
    acc = wmma_bf16(a, b, acc);
  }
  const int col = nt * 16 + (lane & 15);
  const int grp = lane >> 4;
#pragma unroll
  for (int r = 0; r < 8; ++r)
    C[(size_t)(mt * 16 + r + 8 * grp) * NC + col] = acc[r];
}

// ---------------- GAT attention pieces --------------------------------------------
__global__ void k_attn(const float* __restrict__ hx, const float* __restrict__ asrc,
                       const float* __restrict__ adst,
                       float* __restrict__ as_, float* __restrict__ ad_) {
  int idx = blockIdx.x * blockDim.x + threadIdx.x;
  if (idx >= NN * 4) return;
  int n = idx >> 2, h = idx & 3;
  float s = 0.f, d = 0.f;
  const float* hp = hx + (size_t)n * 256 + h * 64;
#pragma unroll 8
  for (int o = 0; o < 64; ++o) { float v = hp[o]; s += v * asrc[h * 64 + o]; d += v * adst[h * 64 + o]; }
  as_[idx] = s; ad_[idx] = d;
}

__device__ __forceinline__ void atomicMaxF(unsigned int* addr, float val) {
  unsigned int u = __float_as_uint(val);
  if ((int)u >= 0) atomicMax((int*)addr, (int)u);
  else             atomicMin(addr, u);
}

__global__ void k_edge_max(const int* __restrict__ eidx, const float* __restrict__ as_,
                           const float* __restrict__ ad_, float* __restrict__ ebuf,
                           unsigned int* __restrict__ maxb) {
  int idx = blockIdx.x * blockDim.x + threadIdx.x;
  if (idx >= (EE + NN) * 4) return;
  int e = idx >> 2, h = idx & 3;
  int s, d;
  if (e < EE) { s = eidx[e]; d = eidx[EE + e]; } else { s = d = e - EE; }
  float v = lrelu(as_[s * 4 + h] + ad_[d * 4 + h]);
  ebuf[idx] = v;
  atomicMaxF(&maxb[d * 4 + h], v);
}

__global__ void k_edge_exp(const int* __restrict__ eidx, float* __restrict__ ebuf,
                           const unsigned int* __restrict__ maxb, float* __restrict__ sumb) {
  int idx = blockIdx.x * blockDim.x + threadIdx.x;
  if (idx >= (EE + NN) * 4) return;
  int e = idx >> 2, h = idx & 3;
  int d = (e < EE) ? eidx[EE + e] : (e - EE);
  float m = __uint_as_float(maxb[d * 4 + h]);
  float ex = __expf(ebuf[idx] - m);
  ebuf[idx] = ex;
  atomicAdd(&sumb[d * 4 + h], ex);
}

__global__ void k_edge_aggr(const int* __restrict__ eidx, const float* __restrict__ ebuf,
                            const float* __restrict__ sumb, const float* __restrict__ hx,
                            float* __restrict__ outacc) {
  int idx = blockIdx.x * blockDim.x + threadIdx.x;
  if (idx >= (EE + NN) * 4) return;
  int e = idx >> 2, h = idx & 3;
  int s, d;
  if (e < EE) { s = eidx[e]; d = eidx[EE + e]; } else { s = d = e - EE; }
  float coeff = ebuf[idx] / (sumb[d * 4 + h] + 1e-16f);
  const float* hp = hx + (size_t)s * 256 + h * 64;
  float* op = outacc + (size_t)d * 256 + h * 64;
#pragma unroll 4
  for (int o = 0; o < 64; ++o) atomicAdd(&op[o], hp[o] * coeff);
}

__global__ void k_fin_concat(const float* __restrict__ outacc, const float* __restrict__ bias,
                             float* __restrict__ xout) {
  int idx = blockIdx.x * blockDim.x + threadIdx.x;
  if (idx >= NN * 256) return;
  xout[idx] = outacc[idx] + bias[idx & 255];
}

__global__ void k_fin_mean(const float* __restrict__ outacc, const float* __restrict__ bias,
                           float* __restrict__ out) {
  int idx = blockIdx.x * blockDim.x + threadIdx.x;
  if (idx >= NN * 64) return;
  int n = idx >> 6, o = idx & 63;
  float s = 0.f;
#pragma unroll
  for (int h = 0; h < 4; ++h) s += outacc[(size_t)n * 256 + h * 64 + o];
  out[idx] = s * 0.25f + bias[o];
}

// ==================================================================================
extern "C" void kernel_launch(void* const* d_in, const int* in_sizes, int n_in,
                              void* d_out, int out_size, void* d_ws, size_t ws_size,
                              hipStream_t stream) {
  const float* x    = (const float*)d_in[0];
  const int*   eidx = (const int*)  d_in[1];
  const float* attr = (const float*)d_in[2];
  const float* w1   = (const float*)d_in[3];
  const float* b1   = (const float*)d_in[4];
  const float* w2   = (const float*)d_in[5];
  const float* b2   = (const float*)d_in[6];
  const float* w3   = (const float*)d_in[7];
  const float* b3   = (const float*)d_in[8];
  const float* root = (const float*)d_in[9];
  const float* nnb  = (const float*)d_in[10];
  const float *gW[4], *gas[4], *gad[4], *gb[4];
  for (int li = 0; li < 4; ++li) {
    gW[li]  = (const float*)d_in[11 + li * 4 + 0];
    gas[li] = (const float*)d_in[11 + li * 4 + 1];
    gad[li] = (const float*)d_in[11 + li * 4 + 2];
    gb[li]  = (const float*)d_in[11 + li * 4 + 3];
  }

  char* ws = (char*)d_ws;
  size_t off = 0;
  auto take = [&](size_t bytes) -> char* {
    char* p = ws + off;
    off = (off + bytes + 255) & ~(size_t)255;
    return p;
  };
  unsigned int* w2p   = (unsigned int*)take((size_t)8  * 64 * 256 * 4);   // 512 KB
  unsigned int* w3p   = (unsigned int*)take((size_t)32 * 64 * 256 * 4);   // 2 MB
  unsigned int* gp[4];
  gp[0] = (unsigned int*)take((size_t)2 * 16 * 256 * 4);
  for (int li = 1; li < 4; ++li) gp[li] = (unsigned int*)take((size_t)8 * 16 * 256 * 4);
  float*          accn   = (float*)take((size_t)NN * 64 * 4);
  float*          xcur   = (float*)take((size_t)MPAD * 256 * 4);
  unsigned short* xbf    = (unsigned short*)take((size_t)MPAD * 256 * 2);
  float*          hx     = (float*)take((size_t)MPAD * 256 * 4);
  float*          as_    = (float*)take((size_t)NN * 4 * 4);
  float*          ad_    = (float*)take((size_t)NN * 4 * 4);
  unsigned int*   maxb   = (unsigned int*)take((size_t)NN * 4 * 4);
  float*          sumb   = (float*)take((size_t)NN * 4 * 4);
  float*          ebuf   = (float*)take((size_t)(EE + NN) * 4 * 4);
  float*          outacc = (float*)take((size_t)NN * 256 * 4);

  const int T = 256;
  auto blks = [](long long n, int t) { return (int)((n + t - 1) / t); };

  // --- pack weights into WMMA B-fragment order (bf16) ---
  k_pack_b<<<blks(8LL  * 64 * 32, T), T, 0, stream>>>(w2, w2p, 256, 1024);
  k_pack_b<<<blks(32LL * 64 * 32, T), T, 0, stream>>>(w3, w3p, 1024, 1024);
  k_pack_b<<<blks(2LL  * 16 * 32, T), T, 0, stream>>>(gW[0], gp[0], 64, 256);
  for (int li = 1; li < 4; ++li)
    k_pack_b<<<blks(8LL * 16 * 32, T), T, 0, stream>>>(gW[li], gp[li], 256, 256);

  // --- NNConv: fused edge MLP + message + scatter ---
  k_fill<<<blks((long long)NN * 64, T), T, 0, stream>>>((unsigned int*)accn, 0u, (size_t)NN * 64);
  k_edge_fused<<<EE / TILE_E, T, 0, stream>>>(x, eidx, attr, w1, b1, w2p, b2, w3p, b3, accn);
  k_root<<<blks((long long)NN * 64, T), T, 0, stream>>>(x, root, nnb, accn, xcur);

  // --- 4 GAT layers ---
  for (int li = 0; li < 4; ++li) {
    const int K = (li == 0) ? 64 : 256;
    k_to_bf16<<<blks((long long)MPAD * K, T), T, 0, stream>>>(xcur, xbf, NN, K);
    k_gemm_nodes<<<blks((long long)(MPAD / 16) * 16, 8), T, 0, stream>>>(xbf, gp[li], hx, K, 256);
    k_attn<<<blks((long long)NN * 4, T), T, 0, stream>>>(hx, gas[li], gad[li], as_, ad_);
    k_fill<<<blks((long long)NN * 4, T), T, 0, stream>>>(maxb, 0xFF800000u, (size_t)NN * 4);
    k_fill<<<blks((long long)NN * 4, T), T, 0, stream>>>((unsigned int*)sumb, 0u, (size_t)NN * 4);
    k_fill<<<blks((long long)NN * 256, T), T, 0, stream>>>((unsigned int*)outacc, 0u, (size_t)NN * 256);
    k_edge_max <<<blks((long long)(EE + NN) * 4, T), T, 0, stream>>>(eidx, as_, ad_, ebuf, maxb);
    k_edge_exp <<<blks((long long)(EE + NN) * 4, T), T, 0, stream>>>(eidx, ebuf, maxb, sumb);
    k_edge_aggr<<<blks((long long)(EE + NN) * 4, T), T, 0, stream>>>(eidx, ebuf, sumb, hx, outacc);
    if (li < 3)
      k_fin_concat<<<blks((long long)NN * 256, T), T, 0, stream>>>(outacc, gb[li], xcur);
    else
      k_fin_mean<<<blks((long long)NN * 64, T), T, 0, stream>>>(outacc, gb[li], (float*)d_out);
  }
  (void)in_sizes; (void)n_in; (void)out_size; (void)ws_size;
}
